// SelfAttentionV3_35081292874395
// MI455X (gfx1250) — matrix operand order repaired and checked by reading, not compiled, not run on hardware
//
#include <hip/hip_runtime.h>
#include <hip/hip_bf16.h>

// ---------------------------------------------------------------------------
// Fused self-attention for MI455X (gfx1250, wave32, WMMA).
//   X[4,4096,512] f32, mask[4,4096,4096] i32, Wqkv[512,1536], bqkv[1536],
//   Wo[512,512], bo[512]  ->  out[4,4096,512] f32
// Pipeline: bf16 convert -> QKV GEMM -> flash attention (online softmax,
// never materializes S x S in HBM) -> output projection GEMM.
// Matmuls: v_wmma_f32_16x16x32_bf16. Tile staging: global_load_async_to_lds
// (ASYNCcnt). Attention processes 32 queries per workgroup to halve K/V
// re-read traffic from L2 and double B-fragment reuse per WMMA.
// ---------------------------------------------------------------------------

#define DEVINL __device__ __forceinline__

typedef __bf16 bf16_t;
typedef __attribute__((ext_vector_type(16))) __bf16 v16bf;
typedef __attribute__((ext_vector_type(8)))  __bf16 v8bf;
typedef __attribute__((ext_vector_type(8)))  float  v8f;

static constexpr int BB = 4;
static constexpr int SS = 4096;
static constexpr int DD = 512;
static constexpr int TD = 3 * DD;      // 1536
static constexpr float NEG_INF = -1.0e20f;

// --- CDNA5 async global->LDS staging (guarded: falls back to plain loads) ---
#if defined(__has_builtin)
#if __has_builtin(__builtin_amdgcn_global_load_async_to_lds_b128) && \
    __has_builtin(__builtin_amdgcn_s_wait_asynccnt)
#define HAVE_ASYNC_LDS 1
#endif
#endif
#ifndef HAVE_ASYNC_LDS
#define HAVE_ASYNC_LDS 0
#endif

#if HAVE_ASYNC_LDS
// Builtin prototype (from hipcc diagnostic): int4* in addrspace(1)/(3).
typedef int v4i_t __attribute__((vector_size(16)));
typedef __attribute__((address_space(1))) v4i_t gas_v4i;
typedef __attribute__((address_space(3))) v4i_t las_v4i;
#endif

// Stage `nrows` x 512 bf16 rows from global (row stride = `stride` elements)
// into LDS (row-major, 512 wide) using the whole 128-thread block.
DEVINL void stage_rows_512(bf16_t* lds_dst, const bf16_t* g_src,
                           size_t stride, int nrows, int tid) {
#if HAVE_ASYNC_LDS
  int chunks = nrows * 64;             // 64 x 16-byte chunks per 1 KB row
  for (int i = tid; i < chunks; i += 128) {
    int row = i >> 6, chunk = i & 63;
    const bf16_t* g = g_src + (size_t)row * stride + chunk * 8;
    bf16_t* l = lds_dst + row * 512 + chunk * 8;
    __builtin_amdgcn_global_load_async_to_lds_b128((gas_v4i*)g, (las_v4i*)l,
                                                   0, 0);
  }
  __builtin_amdgcn_s_wait_asynccnt(0);
#else
  for (int i = tid; i < nrows * 512; i += 128) {
    int row = i >> 9, col = i & 511;
    lds_dst[row * 512 + col] = g_src[(size_t)row * stride + col];
  }
#endif
}

// Concatenate two 8x bf16 16-byte chunks into a 16-element WMMA operand.
DEVINL v16bf frag_cat(const bf16_t* p0, const bf16_t* p1) {
  v8bf lo = *(const v8bf*)p0;
  v8bf hi = *(const v8bf*)p1;
  return __builtin_shufflevector(lo, hi, 0, 1, 2, 3, 4, 5, 6, 7,
                                 8, 9, 10, 11, 12, 13, 14, 15);
}

// A-operand (16x32 bf16, row-major source). `row` points at
// (tile_row_of_this_lane, k0). Lane half h picks the K sub-halves:
//   elems 0..7 -> K = 8h+0..7 ; elems 8..15 -> K = 16+8h+0..7
DEVINL v16bf frag_a(const bf16_t* row, int h) {
  const bf16_t* p = row + 8 * h;
  return frag_cat(p, p + 16);
}

// B-operand (32x16 bf16) from a TRANSPOSED matrix (rows = N, contiguous K).
// `row` points at (n_of_this_lane, k0): elems 0..7 -> K = 8h+0..7 ; elems 8..15 -> K = 16+8h+0..7
DEVINL v16bf frag_b(const bf16_t* row, int h) {
  const bf16_t* p = row + 8 * h;
  return frag_cat(p, p + 16);
}

DEVINL v8f wmma_bf16(v16bf a, v16bf b, v8f c) {
  return __builtin_amdgcn_wmma_f32_16x16x32_bf16(
      false, a, false, b, (short)0, c, false, false);
}

// ---------------------------------------------------------------------------
// Prep: f32 -> bf16 convert (and transposed variant for weights).
// ---------------------------------------------------------------------------
__global__ __launch_bounds__(256) void cvt_f32_bf16(
    const float* __restrict__ src, bf16_t* __restrict__ dst, int n) {
  int i = blockIdx.x * 256 + threadIdx.x;
  if (i < n) dst[i] = (bf16_t)src[i];
}

// src is [Kin x Nout] row-major; dst is [Nout x Kin] row-major (= src^T).
__global__ __launch_bounds__(256) void transpose_cvt(
    const float* __restrict__ src, bf16_t* __restrict__ dst, int Nout, int Kin) {
  int i = blockIdx.x * 256 + threadIdx.x;
  if (i < Nout * Kin) {
    int n = i / Kin, k = i % Kin;
    dst[i] = (bf16_t)src[(size_t)k * Nout + n];
  }
}

// ---------------------------------------------------------------------------
// QKV GEMM: [16384 x 512] @ [512 x 1536] + bias. Block = one 16-row M-tile
// staged once in LDS + 4 N-tiles (one per wave). A-frags from LDS, B-frags
// from global (weights live in L2). Q,K row-major bf16; V transposed [B,D,S].
// ---------------------------------------------------------------------------
__global__ __launch_bounds__(128) void qkv_gemm(
    const bf16_t* __restrict__ Xb, const bf16_t* __restrict__ Wt,
    const float* __restrict__ bias, bf16_t* __restrict__ Qb,
    bf16_t* __restrict__ Kb, bf16_t* __restrict__ Vt) {
  __shared__ bf16_t At[16 * 512];
  const int NG = TD / 64;             // 24 groups of 4 N-tiles
  int wave = threadIdx.x >> 5;
  int lane = threadIdx.x & 31;
  int ln = lane & 15, h = lane >> 4;
  int tm = blockIdx.x / NG;
  int tn = (blockIdx.x % NG) * 4 + wave;

  stage_rows_512(At, Xb + (size_t)(tm * 16) * DD, DD, 16, threadIdx.x);
  __syncthreads();

  const bf16_t* arow = At + (size_t)ln * DD;
  const bf16_t* brow = Wt + (size_t)(tn * 16 + ln) * DD;

  v8f acc = {};
#pragma unroll 4
  for (int k0 = 0; k0 < DD; k0 += 32) {
    v16bf a = frag_a(arow + k0, h);
    v16bf b = frag_b(brow + k0, h);
    acc = wmma_bf16(a, b, acc);
  }

  int col = tn * 16 + ln;             // C column of this lane
  float bv = bias[col];
#pragma unroll
  for (int r = 0; r < 8; ++r) {
    int mm = tm * 16 + r + 8 * h;     // C row (per ISA C/D layout)
    float v = acc[r] + bv;
    if (col < DD) {
      Qb[(size_t)mm * DD + col] = (bf16_t)v;
    } else if (col < 2 * DD) {
      Kb[(size_t)mm * DD + (col - DD)] = (bf16_t)v;
    } else {
      int bidx = mm / SS, s = mm % SS;
      Vt[((size_t)bidx * DD + (col - 2 * DD)) * SS + s] = (bf16_t)v;
    }
  }
}

// ---------------------------------------------------------------------------
// Flash attention: one workgroup (4 waves) per (batch, 32-query block).
// Streams keys in blocks of 64; online softmax; O kept in registers
// (each wave owns a 128-wide slice of D, for both 16-query tiles).
// K-fragments feed 2 WMMAs (both q-tiles); V-fragments likewise.
// ---------------------------------------------------------------------------
__global__ __launch_bounds__(128) void attn_kernel(
    const bf16_t* __restrict__ Qb, const bf16_t* __restrict__ Kb,
    const bf16_t* __restrict__ Vt, const int* __restrict__ mask,
    bf16_t* __restrict__ AV) {
  __shared__ bf16_t Qt[32 * DD];      // 32 KB query tile (A-operand source)
  __shared__ float  Pf[32 * 64];      // masked logits (f32)
  __shared__ bf16_t Pb[32 * 64];      // softmax probs (bf16 A-operand)
  __shared__ float  red[32 * 4];      // per-row partial reductions
  __shared__ float  mrow[32], lrow[32], srow[32];

  const int bidx = blockIdx.y;
  const int q0 = blockIdx.x * 32;
  const int tid = threadIdx.x;
  const int wave = tid >> 5, lane = tid & 31;
  const int ln = lane & 15, h = lane >> 4;
  const int srid = tid >> 2, sseg = tid & 3;      // softmax: row / 16-col seg
  const float rs = 0.044194173824159216f;         // 1/sqrt(512)

  stage_rows_512(Qt, Qb + ((size_t)bidx * SS + q0) * DD, DD, 32, tid);
  if (tid < 32) { mrow[tid] = -3.0e38f; lrow[tid] = 0.0f; }

  v8f o[2][8];
#pragma unroll
  for (int qt = 0; qt < 2; ++qt)
#pragma unroll
    for (int j = 0; j < 8; ++j) o[qt][j] = (v8f){};
  __syncthreads();

  const size_t mask_base = ((size_t)bidx * SS + q0) * SS;

  for (int k0 = 0; k0 < SS; k0 += 64) {
    int kw = k0 + wave * 16;          // this wave's 16-key tile

    // Prefetch next key block's mask rows (mask stream dominates HBM BW).
    __builtin_prefetch(mask + mask_base + (size_t)(8 * h) * SS + k0 + 64 + ln,
                       0, 1);

    // ---- logits: two 16x16 tiles (one per q-tile), shared K-fragment ----
    v8f s0 = {}, s1 = {};
    const bf16_t* qrow0 = Qt + (size_t)ln * DD;
    const bf16_t* qrow1 = Qt + (size_t)(16 + ln) * DD;
    const bf16_t* krow = Kb + ((size_t)bidx * SS + kw + ln) * DD;
#pragma unroll 4
    for (int kk = 0; kk < DD; kk += 32) {
      v16bf b = frag_b(krow + kk, h);           // loaded once, used twice
      s0 = wmma_bf16(frag_a(qrow0 + kk, h), b, s0);
      s1 = wmma_bf16(frag_a(qrow1 + kk, h), b, s1);
    }

    // ---- mask + stage logits to LDS ----
#pragma unroll
    for (int r = 0; r < 8; ++r) {
      int mq0 = r + 8 * h, mq1 = 16 + mq0;
      int mv0 = mask[mask_base + (size_t)mq0 * SS + kw + ln];
      int mv1 = mask[mask_base + (size_t)mq1 * SS + kw + ln];
      float v0 = s0[r] * rs, v1 = s1[r] * rs;
      Pf[mq0 * 64 + wave * 16 + ln] = (mv0 == 0) ? NEG_INF : v0;
      Pf[mq1 * 64 + wave * 16 + ln] = (mv1 == 0) ? NEG_INF : v1;
    }
    __syncthreads();

    // ---- online softmax: 32 rows x 4 segments of 16 ----
    {  // partial max over this thread's 16 logits
      const float* prow = Pf + srid * 64 + sseg * 16;
      float mx = prow[0];
#pragma unroll
      for (int j = 1; j < 16; ++j) mx = fmaxf(mx, prow[j]);
      red[srid * 4 + sseg] = mx;
    }
    __syncthreads();
    if (tid < 32) {                   // reduce partials, update running max
      float mo = mrow[tid];
      float mx = mo;
#pragma unroll
      for (int s = 0; s < 4; ++s) mx = fmaxf(mx, red[tid * 4 + s]);
      srow[tid] = __expf(mo - mx);
      mrow[tid] = mx;
    }
    __syncthreads();
    {  // exp + bf16 probs + partial sums
      float mx = mrow[srid];
      const float* prow = Pf + srid * 64 + sseg * 16;
      bf16_t* pb = Pb + srid * 64 + sseg * 16;
      float sum = 0.0f;
#pragma unroll
      for (int j = 0; j < 16; ++j) {
        float p = __expf(prow[j] - mx);
        pb[j] = (bf16_t)p;
        sum += p;
      }
      red[srid * 4 + sseg] = sum;
    }
    __syncthreads();
    if (tid < 32) {                   // fold partial sums into running sum
      float sum = lrow[tid] * srow[tid];
#pragma unroll
      for (int s = 0; s < 4; ++s) sum += red[tid * 4 + s];
      lrow[tid] = sum;
    }
    __syncthreads();

    // ---- rescale O accumulators by exp(m_old - m_new) ----
#pragma unroll
    for (int qt = 0; qt < 2; ++qt) {
      float scr[8];
#pragma unroll
      for (int r = 0; r < 8; ++r) scr[r] = srow[qt * 16 + r + 8 * h];
#pragma unroll
      for (int j = 0; j < 8; ++j)
#pragma unroll
        for (int r = 0; r < 8; ++r) o[qt][j][r] *= scr[r];
    }

    // ---- O += P[32 x 64] @ V[64 x 128-slice], shared V-fragment ----
#pragma unroll
    for (int half = 0; half < 2; ++half) {
      v16bf a0 = frag_a(Pb + ln * 64 + half * 32, h);
      v16bf a1 = frag_a(Pb + (16 + ln) * 64 + half * 32, h);
#pragma unroll
      for (int j = 0; j < 8; ++j) {
        int d = wave * 128 + j * 16 + ln;
        const bf16_t* vrow =
            Vt + ((size_t)bidx * DD + d) * SS + k0 + half * 32;
        v16bf b = frag_b(vrow, h);              // loaded once, used twice
        o[0][j] = wmma_bf16(a0, b, o[0][j]);
        o[1][j] = wmma_bf16(a1, b, o[1][j]);
      }
    }
    __syncthreads();
  }

  // ---- finalize: divide by running sum, write bf16 attn*V ----
#pragma unroll
  for (int qt = 0; qt < 2; ++qt)
#pragma unroll
    for (int j = 0; j < 8; ++j) {
      int d = wave * 128 + j * 16 + ln;
#pragma unroll
      for (int r = 0; r < 8; ++r) {
        int mq = qt * 16 + r + 8 * h;
        float v = o[qt][j][r] / lrow[mq];
        AV[((size_t)bidx * SS + q0 + mq) * DD + d] = (bf16_t)v;
      }
    }
}

// ---------------------------------------------------------------------------
// Output projection: [16384 x 512] @ [512 x 512] + bo -> f32 out.
// Same LDS-staged A-tile scheme as qkv_gemm.
// ---------------------------------------------------------------------------
__global__ __launch_bounds__(128) void out_gemm(
    const bf16_t* __restrict__ AV, const bf16_t* __restrict__ Wot,
    const float* __restrict__ bo, float* __restrict__ out) {
  __shared__ bf16_t At[16 * 512];
  const int NG = DD / 64;             // 8 groups of 4 N-tiles
  int wave = threadIdx.x >> 5;
  int lane = threadIdx.x & 31;
  int ln = lane & 15, h = lane >> 4;
  int tm = blockIdx.x / NG;
  int tn = (blockIdx.x % NG) * 4 + wave;

  stage_rows_512(At, AV + (size_t)(tm * 16) * DD, DD, 16, threadIdx.x);
  __syncthreads();

  const bf16_t* arow = At + (size_t)ln * DD;
  const bf16_t* brow = Wot + (size_t)(tn * 16 + ln) * DD;

  v8f acc = {};
#pragma unroll 4
  for (int k0 = 0; k0 < DD; k0 += 32) {
    v16bf a = frag_a(arow + k0, h);
    v16bf b = frag_b(brow + k0, h);
    acc = wmma_bf16(a, b, acc);
  }

  int col = tn * 16 + ln;
  float bv = bo[col];
#pragma unroll
  for (int r = 0; r < 8; ++r) {
    int mm = tm * 16 + r + 8 * h;
    out[(size_t)mm * DD + col] = acc[r] + bv;
  }
}

// ---------------------------------------------------------------------------
// Host-side launch.
// ---------------------------------------------------------------------------
extern "C" void kernel_launch(void* const* d_in, const int* in_sizes, int n_in,
                              void* d_out, int out_size, void* d_ws,
                              size_t ws_size, hipStream_t stream) {
  (void)in_sizes; (void)n_in; (void)out_size; (void)ws_size;
  const float* X    = (const float*)d_in[0];
  const int*   mask = (const int*)d_in[1];
  const float* Wqkv = (const float*)d_in[2];
  const float* bqkv = (const float*)d_in[3];
  const float* Wo   = (const float*)d_in[4];
  const float* bo   = (const float*)d_in[5];
  float* out = (float*)d_out;

  // Workspace carve-up (bf16 buffers, all multiples of 512 KB => aligned).
  const size_t nM = (size_t)BB * SS * DD;     // 8,388,608 elements
  char* w = (char*)d_ws;
  bf16_t* Xb    = (bf16_t*)w;  w += nM * sizeof(bf16_t);
  bf16_t* Wqkvt = (bf16_t*)w;  w += (size_t)TD * DD * sizeof(bf16_t);
  bf16_t* Wot   = (bf16_t*)w;  w += (size_t)DD * DD * sizeof(bf16_t);
  bf16_t* Qb    = (bf16_t*)w;  w += nM * sizeof(bf16_t);
  bf16_t* Kb    = (bf16_t*)w;  w += nM * sizeof(bf16_t);
  bf16_t* Vt    = (bf16_t*)w;  w += nM * sizeof(bf16_t);
  bf16_t* AV    = (bf16_t*)w;  w += nM * sizeof(bf16_t);
  // total ~85 MB of workspace

  // 1) bf16 conversions / weight transposes
  cvt_f32_bf16<<<(int)(nM / 256), 256, 0, stream>>>(X, Xb, (int)nM);
  transpose_cvt<<<(TD * DD) / 256, 256, 0, stream>>>(Wqkv, Wqkvt, TD, DD);
  transpose_cvt<<<(DD * DD) / 256, 256, 0, stream>>>(Wo, Wot, DD, DD);

  // 2) QKV projection: 1024 M-tiles x 24 N-groups
  qkv_gemm<<<(BB * SS / 16) * (TD / 64), 128, 0, stream>>>(
      Xb, Wqkvt, bqkv, Qb, Kb, Vt);

  // 3) Flash attention: one block per (32-query block, batch)
  attn_kernel<<<dim3(SS / 32, BB), 128, 0, stream>>>(Qb, Kb, Vt, mask, AV);

  // 4) Output projection: 1024 M-tiles x 8 N-groups
  out_gemm<<<(BB * SS / 16) * (DD / 64), 128, 0, stream>>>(
      AV, Wot, bo, out);
}
